// CostVolumeDepthPredictor_21466246546099
// MI455X (gfx1250) — matrix operand order repaired and checked by reading, not compiled, not run on hardware
//
#include <hip/hip_runtime.h>
#include <math.h>

#define NV    4
#define CCH   64
#define DN    64
#define LH    64
#define LW    96
#define LHW   (LH*LW)          // 6144
#define FH    256
#define FW    384
#define FHW   (FH*FW)          // 98304
#define INV_MIN_D 0.1f         // 1/MAX_DEPTH
#define INV_MAX_D 2.0f         // 1/MIN_DEPTH
#define INTERVAL  (1.9f/63.0f)

typedef __attribute__((ext_vector_type(16))) _Float16 v16h;
typedef __attribute__((ext_vector_type(8)))  _Float16 h8;
typedef __attribute__((ext_vector_type(8)))  float    v8f;

__device__ __forceinline__ float gelu_f(float v) {
  return 0.5f * v * (1.0f + erff(v * 0.70710678118654752f));
}

// A-fragment (16x32 f16, row-major source, row base pointer given):
// element j -> k = kb + half*16 + j  => one contiguous 16-half run
__device__ __forceinline__ v16h load_a_frag(const _Float16* __restrict__ row,
                                            int kb, int half) {
  h8 a0 = *(const h8*)(row + kb + half * 16);
  h8 a1 = *(const h8*)(row + kb + half * 16 + 8);
  return __builtin_shufflevector(a0, a1, 0,1,2,3,4,5,6,7,8,9,10,11,12,13,14,15);
}

// B-fragment (32x16): lane holds column nc, element j -> k = kb + half*16 + j
__device__ __forceinline__ v16h load_b_frag(const float* __restrict__ X,
                                            int kb, int half, int nc, int P) {
  v16h b;
  #pragma unroll
  for (int j = 0; j < 16; ++j)
    b[j] = (_Float16)X[(size_t)(kb + half * 16 + j) * P + nc];
  return b;
}

// ---------------------------------------------------------------------------
// Weight repack kernels (run once, before the view loop)
// ---------------------------------------------------------------------------
__global__ void pack_w_f16_k(const float* __restrict__ w, _Float16* __restrict__ out, int n) {
  int i = blockIdx.x * blockDim.x + threadIdx.x;
  if (i < n) out[i] = (_Float16)w[i];
}

// fr_conv_w (64, 67, 3, 3) -> Wt[(tap*64 + m)*96 + k], zero pad k>=67
__global__ void pack_conv_w_k(const float* __restrict__ w, _Float16* __restrict__ out) {
  int i = blockIdx.x * blockDim.x + threadIdx.x;
  if (i >= 9 * 64 * 96) return;
  int k = i % 96;
  int t = i / 96;
  int m = t % 64;
  int tap = t / 64;
  out[i] = (k < 67) ? (_Float16)w[(m * 67 + k) * 9 + tap] : (_Float16)0.0f;
}

// ---------------------------------------------------------------------------
// K0: compose per-source-view projection: p = A*[x,y,1]*depth + t
// ---------------------------------------------------------------------------
__device__ __forceinline__ void mat3mul(const float* A, const float* B, float* C) {
  for (int i = 0; i < 3; ++i)
    for (int j = 0; j < 3; ++j)
      C[i*3+j] = A[i*3+0]*B[0*3+j] + A[i*3+1]*B[1*3+j] + A[i*3+2]*B[2*3+j];
}

__global__ void proj_k(const float* __restrict__ Kall, const float* __restrict__ Eall,
                       int r, float* __restrict__ proj) {
  if (threadIdx.x != 0 || blockIdx.x != 0) return;
  float scl[3] = {(float)LW, (float)LH, 1.0f};
  float Kr[9];
  for (int i = 0; i < 3; ++i)
    for (int j = 0; j < 3; ++j)
      Kr[i*3+j] = Kall[r*9 + i*3 + j] * scl[i];
  float det = Kr[0]*(Kr[4]*Kr[8]-Kr[5]*Kr[7])
            - Kr[1]*(Kr[3]*Kr[8]-Kr[5]*Kr[6])
            + Kr[2]*(Kr[3]*Kr[7]-Kr[4]*Kr[6]);
  float id = 1.0f / det;
  float iK[9] = {
    (Kr[4]*Kr[8]-Kr[5]*Kr[7])*id, (Kr[2]*Kr[7]-Kr[1]*Kr[8])*id, (Kr[1]*Kr[5]-Kr[2]*Kr[4])*id,
    (Kr[5]*Kr[6]-Kr[3]*Kr[8])*id, (Kr[0]*Kr[8]-Kr[2]*Kr[6])*id, (Kr[2]*Kr[3]-Kr[0]*Kr[5])*id,
    (Kr[3]*Kr[7]-Kr[4]*Kr[6])*id, (Kr[1]*Kr[6]-Kr[0]*Kr[7])*id, (Kr[0]*Kr[4]-Kr[1]*Kr[3])*id };
  float Rr[9], tr[3];
  for (int i = 0; i < 3; ++i) {
    for (int j = 0; j < 3; ++j) Rr[i*3+j] = Eall[r*16 + i*4 + j];
    tr[i] = Eall[r*16 + i*4 + 3];
  }
  float RrK[9]; mat3mul(Rr, iK, RrK);
  int b = 0;
  for (int v = 0; v < NV; ++v) {
    if (v == r) continue;
    float Rsrc[9], tsrc[3];
    for (int i = 0; i < 3; ++i) {
      for (int j = 0; j < 3; ++j) Rsrc[i*3+j] = Eall[v*16 + i*4 + j];
      tsrc[i] = Eall[v*16 + i*4 + 3];
    }
    float Rs[9], ts[3];
    for (int i = 0; i < 3; ++i)
      for (int j = 0; j < 3; ++j) Rs[i*3+j] = Rsrc[j*3+i];
    for (int i = 0; i < 3; ++i)
      ts[i] = -(Rs[i*3]*tsrc[0] + Rs[i*3+1]*tsrc[1] + Rs[i*3+2]*tsrc[2]);
    float Ks[9];
    for (int i = 0; i < 3; ++i)
      for (int j = 0; j < 3; ++j) Ks[i*3+j] = Kall[v*9 + i*3 + j] * scl[i];
    float T1[9]; mat3mul(Rs, RrK, T1);
    float A[9];  mat3mul(Ks, T1, A);
    float u[3], t[3];
    for (int i = 0; i < 3; ++i)
      u[i] = Rs[i*3]*tr[0] + Rs[i*3+1]*tr[1] + Rs[i*3+2]*tr[2] + ts[i];
    for (int i = 0; i < 3; ++i)
      t[i] = Ks[i*3]*u[0] + Ks[i*3+1]*u[1] + Ks[i*3+2]*u[2];
    for (int k = 0; k < 9; ++k) proj[b*12 + k] = A[k];
    for (int i = 0; i < 3; ++i) proj[b*12 + 9 + i] = t[i];
    ++b;
  }
}

// ---------------------------------------------------------------------------
// K1: fused plane-sweep warp + cosine cost + visibility aggregation
// ---------------------------------------------------------------------------
__global__ void cost_volume_k(const float* __restrict__ low, const float* __restrict__ proj,
                              int r, float* __restrict__ agg) {
  int idx = blockIdx.x * blockDim.x + threadIdx.x;
  if (idx >= DN * LHW) return;
  int d = idx / LHW;
  int p = idx - d * LHW;
  int h = p / LW;
  int w = p - h * LW;
  float dep = 1.0f / (INV_MIN_D + (float)d * INTERVAL);
  float fx = (float)w, fy = (float)h;

  int   off[3][4];
  float wgt[3][4];
  float vis[3];
  const float* sbase[3];
  #pragma unroll
  for (int b = 0; b < 3; ++b) {
    const float* M = proj + b * 12;
    float X = (M[0]*fx + M[1]*fy + M[2]) * dep + M[9];
    float Y = (M[3]*fx + M[4]*fy + M[5]) * dep + M[10];
    float Z = (M[6]*fx + M[7]*fy + M[8]) * dep + M[11];
    float den = fmaxf(Z, 1e-6f);
    float sx = X / den, sy = Y / den;
    vis[b] = (Z > 1e-6f && sx >= 0.0f && sx <= (float)(LW-1) &&
              sy >= 0.0f && sy <= (float)(LH-1)) ? 1.0f : 0.0f;
    float x0f = floorf(sx), y0f = floorf(sy);
    float wx = sx - x0f, wy = sy - y0f;
    int x0 = (int)x0f, y0 = (int)y0f;
    int xs[4] = {x0, x0+1, x0,   x0+1};
    int ys[4] = {y0, y0,   y0+1, y0+1};
    float wt[4] = {(1.f-wx)*(1.f-wy), wx*(1.f-wy), (1.f-wx)*wy, wx*wy};
    #pragma unroll
    for (int t = 0; t < 4; ++t) {
      bool ok = (unsigned)xs[t] < (unsigned)LW && (unsigned)ys[t] < (unsigned)LH;
      off[b][t] = ok ? (ys[t]*LW + xs[t]) : 0;
      wgt[b][t] = ok ? wt[t] : 0.0f;
    }
    int sv = b + (b >= r ? 1 : 0);
    sbase[b] = low + (size_t)sv * CCH * LHW;
  }

  const float* ref = low + (size_t)r * CCH * LHW + p;
  float nr = 0.0f;
  float dot[3] = {0.f, 0.f, 0.f};
  float nw[3]  = {0.f, 0.f, 0.f};
  for (int c = 0; c < CCH; ++c) {
    float rv = ref[c * LHW];
    nr += rv * rv;
    #pragma unroll
    for (int b = 0; b < 3; ++b) {
      const float* sb = sbase[b] + c * LHW;
      float v = wgt[b][0]*sb[off[b][0]] + wgt[b][1]*sb[off[b][1]]
              + wgt[b][2]*sb[off[b][2]] + wgt[b][3]*sb[off[b][3]];
      dot[b] += rv * v;
      nw[b]  += v * v;
    }
  }
  float num = 0.0f, den2 = 0.0f;
  float snr = fmaxf(sqrtf(nr), 1e-8f);
  #pragma unroll
  for (int b = 0; b < 3; ++b) {
    if (vis[b] > 0.0f) {
      float cosv = dot[b] / (snr * fmaxf(sqrtf(nw[b]), 1e-8f));
      num += 1.0f - cosv;
      den2 += 1.0f;
    }
  }
  agg[idx] = num / fmaxf(den2, 1.0f);
}

// ---------------------------------------------------------------------------
// K2: depthwise 3x3 conv, pad 1, no bias
// ---------------------------------------------------------------------------
__global__ void dwconv3_k(const float* __restrict__ x, const float* __restrict__ wts,
                          float* __restrict__ out, int H, int W, int total) {
  int idx = blockIdx.x * blockDim.x + threadIdx.x;
  if (idx >= total) return;
  int HW = H * W;
  int c = idx / HW;
  int p = idx - c * HW;
  int h = p / W;
  int wx = p - h * W;
  const float* xp = x + (size_t)c * HW;
  const float* wp = wts + c * 9;
  float s = 0.0f;
  #pragma unroll
  for (int dy = -1; dy <= 1; ++dy)
    #pragma unroll
    for (int dx = -1; dx <= 1; ++dx) {
      int yy = h + dy, xx = wx + dx;
      if ((unsigned)yy < (unsigned)H && (unsigned)xx < (unsigned)W)
        s += wp[(dy+1)*3 + (dx+1)] * xp[yy*W + xx];
    }
  out[idx] = s;
}

// ---------------------------------------------------------------------------
// K3a: pointwise GEMM (K=64, Cout=64) — one wave per 16-col tile, 4 M-tiles,
// B-fragment reused across 4 WMMAs. out = W@X + bias
// ---------------------------------------------------------------------------
__global__ void pw_gemm16_k(const _Float16* __restrict__ Wf, const float* __restrict__ X,
                            const float* __restrict__ bias, float* __restrict__ out, int P) {
  int wid  = (blockIdx.x * blockDim.x + threadIdx.x) >> 5;   // N-tile
  int lane = threadIdx.x & 31;
  int half = lane >> 4;
  int l15  = lane & 15;
  int nc = (wid << 4) + l15;
  v8f acc[4] = {v8f{}, v8f{}, v8f{}, v8f{}};
  #pragma unroll
  for (int kb = 0; kb < 64; kb += 32) {
    v16h b = load_b_frag(X, kb, half, nc, P);
    #pragma unroll
    for (int mt = 0; mt < 4; ++mt) {
      v16h a = load_a_frag(Wf + (size_t)(mt * 16 + l15) * 64, kb, half);
      acc[mt] = __builtin_amdgcn_wmma_f32_16x16x32_f16(false, a, false, b,
                                                       (short)0, acc[mt], false, false);
    }
  }
  #pragma unroll
  for (int mt = 0; mt < 4; ++mt)
    #pragma unroll
    for (int i = 0; i < 8; ++i) {
      int m = mt * 16 + half * 8 + i;
      out[(size_t)m * P + nc] = acc[mt][i] + bias[m];
    }
}

// K3b: logits GEMM: out = W@X + bias - sub0 - sub1
__global__ void pw_logits16_k(const _Float16* __restrict__ Wf, const float* __restrict__ X,
                              const float* __restrict__ bias, float* __restrict__ out,
                              const float* __restrict__ sub0, const float* __restrict__ sub1,
                              int P) {
  int wid  = (blockIdx.x * blockDim.x + threadIdx.x) >> 5;
  int lane = threadIdx.x & 31;
  int half = lane >> 4;
  int l15  = lane & 15;
  int nc = (wid << 4) + l15;
  v8f acc[4] = {v8f{}, v8f{}, v8f{}, v8f{}};
  #pragma unroll
  for (int kb = 0; kb < 64; kb += 32) {
    v16h b = load_b_frag(X, kb, half, nc, P);
    #pragma unroll
    for (int mt = 0; mt < 4; ++mt) {
      v16h a = load_a_frag(Wf + (size_t)(mt * 16 + l15) * 64, kb, half);
      acc[mt] = __builtin_amdgcn_wmma_f32_16x16x32_f16(false, a, false, b,
                                                       (short)0, acc[mt], false, false);
    }
  }
  #pragma unroll
  for (int mt = 0; mt < 4; ++mt)
    #pragma unroll
    for (int i = 0; i < 8; ++i) {
      int m = mt * 16 + half * 8 + i;
      size_t o = (size_t)m * P + nc;
      out[o] = acc[mt][i] + bias[m] - sub0[o] - sub1[o];
    }
}

// ---------------------------------------------------------------------------
// K4/K5: group norm stats (one block per group) + apply+GELU+residual
// ---------------------------------------------------------------------------
__global__ void gn_stats_k(const float* __restrict__ x, float* __restrict__ stats,
                           int chPerG, int HW) {
  int g = blockIdx.x;
  int tid = threadIdx.x;
  long n = (long)chPerG * HW;
  const float* base = x + (size_t)g * chPerG * HW;
  float s = 0.0f, s2 = 0.0f;
  for (long i = tid; i < n; i += blockDim.x) {
    float v = base[i];
    s += v; s2 += v * v;
  }
  __shared__ float sh[512];
  sh[tid] = s; sh[256 + tid] = s2;
  __syncthreads();
  for (int o = 128; o > 0; o >>= 1) {
    if (tid < o) { sh[tid] += sh[tid + o]; sh[256+tid] += sh[256+tid+o]; }
    __syncthreads();
  }
  if (tid == 0) {
    float fn = (float)n;
    float mean = sh[0] / fn;
    float var  = sh[256] / fn - mean * mean;
    stats[2*g]   = mean;
    stats[2*g+1] = rsqrtf(var + 1e-5f);
  }
}

__global__ void gn_gelu_res_k(const float* __restrict__ x, const float* __restrict__ stats,
                              const float* __restrict__ gamma, const float* __restrict__ beta,
                              const float* __restrict__ res, float* __restrict__ out,
                              int HW, int chPerG, int total) {
  int idx = blockIdx.x * blockDim.x + threadIdx.x;
  if (idx >= total) return;
  int c = idx / HW;
  int g = c / chPerG;
  float v = (x[idx] - stats[2*g]) * stats[2*g+1] * gamma[c] + beta[c];
  out[idx] = gelu_f(v) + res[idx];
}

// ---------------------------------------------------------------------------
// K6: softmax over D + shifted moments -> inv_low, inv_unc_low, pdf_max
// ---------------------------------------------------------------------------
__global__ void softmax_stats_k(const float* __restrict__ logits, float* __restrict__ maps) {
  int p = blockIdx.x * blockDim.x + threadIdx.x;
  if (p >= LHW) return;
  float mx = -3.4e38f;
  for (int d = 0; d < DN; ++d) mx = fmaxf(mx, logits[d * LHW + p]);
  float S0 = 0.f, S1 = 0.f, S2 = 0.f;
  for (int d = 0; d < DN; ++d) {
    float bin = INV_MIN_D + (float)d * INTERVAL;
    float e = expf(logits[d * LHW + p] - mx);
    S0 += e; S1 += e * bin; S2 += e * bin * bin;
  }
  float mean = S1 / S0;
  float var  = fmaxf(S2 / S0 - mean * mean, 0.0f);
  maps[p]          = mean;
  maps[LHW + p]    = fmaxf(sqrtf(var), 1e-6f);
  maps[2*LHW + p]  = 1.0f / S0;
}

// ---------------------------------------------------------------------------
// K8: fullres 3x3 conv (67->64, K padded to 96) as implicit GEMM.
// One wave per 16-col tile, 4 M-tiles, B reused 4x; fused bias+GELU.
// Wt layout: [(tap*64 + m)*96 + k] (f16, pre-packed).
// ---------------------------------------------------------------------------
__global__ void conv3x3_gemm16_k(const float* __restrict__ feat, const float* __restrict__ img,
                                 const _Float16* __restrict__ Wt, const float* __restrict__ bias,
                                 float* __restrict__ out) {
  int wid  = (blockIdx.x * blockDim.x + threadIdx.x) >> 5;
  int lane = threadIdx.x & 31;
  int half = lane >> 4;
  int l15  = lane & 15;
  int p0 = (wid << 4) + l15;
  int yy = p0 / FW;
  int xx = p0 - yy * FW;
  v8f acc[4] = {v8f{}, v8f{}, v8f{}, v8f{}};
  for (int tap = 0; tap < 9; ++tap) {
    int dy = tap / 3 - 1, dx = tap % 3 - 1;
    int py = yy + dy, px = xx + dx;
    bool pv = (unsigned)py < (unsigned)FH && (unsigned)px < (unsigned)FW;
    int sp = py * FW + px;
    #pragma unroll
    for (int kb = 0; kb < 96; kb += 32) {
      v16h b;
      #pragma unroll
      for (int j = 0; j < 16; ++j) {
        int kx = kb + half * 16 + j;
        float bv = 0.0f;
        if (pv && kx < 67)
          bv = (kx < 64) ? feat[(size_t)kx * FHW + sp] : img[(size_t)(kx - 64) * FHW + sp];
        b[j] = (_Float16)bv;
      }
      #pragma unroll
      for (int mt = 0; mt < 4; ++mt) {
        const _Float16* row = Wt + (size_t)(tap * 64 + mt * 16 + l15) * 96;
        v16h a = load_a_frag(row, kb, half);
        acc[mt] = __builtin_amdgcn_wmma_f32_16x16x32_f16(false, a, false, b,
                                                         (short)0, acc[mt], false, false);
      }
    }
  }
  #pragma unroll
  for (int mt = 0; mt < 4; ++mt)
    #pragma unroll
    for (int i = 0; i < 8; ++i) {
      int m = mt * 16 + half * 8 + i;
      out[(size_t)m * FHW + p0] = gelu_f(acc[mt][i] + bias[m]);
    }
}

// ---------------------------------------------------------------------------
// K10: heads — fused bilinear upsample of 3 lowres maps + 1x1 heads
// ---------------------------------------------------------------------------
__device__ __forceinline__ float bilin_low(const float* m, int y0, int x0, int y1, int x1,
                                           float wy, float wx) {
  return m[y0*LW+x0]*(1.f-wy)*(1.f-wx) + m[y0*LW+x1]*(1.f-wy)*wx
       + m[y1*LW+x0]*wy*(1.f-wx)       + m[y1*LW+x1]*wy*wx;
}

__global__ void heads_k(const float* __restrict__ y, const float* __restrict__ maps,
                        const float* __restrict__ res_w, const float* __restrict__ res_b,
                        const float* __restrict__ conf_w, const float* __restrict__ conf_b,
                        float* __restrict__ out, int r) {
  int p = blockIdx.x * blockDim.x + threadIdx.x;
  if (p >= FHW) return;
  int oy = p / FW, ox = p - oy * FW;
  float fy = (float)oy * (float)(LH - 1) / (float)(FH - 1);
  float fx = (float)ox * (float)(LW - 1) / (float)(FW - 1);
  int y0 = (int)floorf(fy), x0 = (int)floorf(fx);
  int y1 = min(y0 + 1, LH - 1), x1 = min(x0 + 1, LW - 1);
  float wy = fy - (float)y0, wx = fx - (float)x0;
  float invF  = bilin_low(maps,         y0, x0, y1, x1, wy, wx);
  float uncF  = bilin_low(maps + LHW,   y0, x0, y1, x1, wy, wx);
  float pmaxF = bilin_low(maps + 2*LHW, y0, x0, y1, x1, wy, wx);
  float s1 = res_b[0], s2 = conf_b[0];
  for (int c = 0; c < CCH; ++c) {
    float v = y[(size_t)c * FHW + p];
    s1 += res_w[c] * v;
    s2 += conf_w[c] * v;
  }
  float resid = tanhf(s1) * INTERVAL;
  float invd  = fminf(fmaxf(invF + resid, INV_MIN_D), INV_MAX_D);
  float depth = 1.0f / invd;
  float unc   = fmaxf(uncF / fmaxf(invd * invd, 1e-6f), 0.01f);
  float sig   = 1.0f / (1.0f + expf(-s2));
  float conf  = sig * pmaxF / (1.0f + uncF / INTERVAL);
  out[(size_t)r * FHW + p]          = depth;
  out[(size_t)(NV + r) * FHW + p]   = unc;
  out[(size_t)(2*NV + r) * FHW + p] = conf;
}

// ---------------------------------------------------------------------------
extern "C" void kernel_launch(void* const* d_in, const int* in_sizes, int n_in,
                              void* d_out, int out_size, void* d_ws, size_t ws_size,
                              hipStream_t stream) {
  (void)in_sizes; (void)n_in; (void)out_size; (void)ws_size;
  const float* low     = (const float*)d_in[0];
  const float* full    = (const float*)d_in[1];
  const float* imgs    = (const float*)d_in[2];
  const float* Kin     = (const float*)d_in[3];
  const float* Ein     = (const float*)d_in[4];
  const float* dp_w    = (const float*)d_in[5];
  const float* dp_b    = (const float*)d_in[6];
  const float* cr_dw   = (const float*)d_in[7];
  const float* cr_pw_w = (const float*)d_in[8];
  const float* cr_pw_b = (const float*)d_in[9];
  const float* cr_gn_g = (const float*)d_in[10];
  const float* cr_gn_b = (const float*)d_in[11];
  const float* fr_cw   = (const float*)d_in[12];
  const float* fr_cb   = (const float*)d_in[13];
  const float* fr_dw   = (const float*)d_in[14];
  const float* fr_pw_w = (const float*)d_in[15];
  const float* fr_pw_b = (const float*)d_in[16];
  const float* fr_gn_g = (const float*)d_in[17];
  const float* fr_gn_b = (const float*)d_in[18];
  const float* res_w   = (const float*)d_in[19];
  const float* res_b   = (const float*)d_in[20];
  const float* conf_w  = (const float*)d_in[21];
  const float* conf_b  = (const float*)d_in[22];
  float* outp = (float*)d_out;

  // ---- workspace layout ----
  // f16 weight region first (16B aligned; 75776 halfs = 151552 bytes, already
  // a multiple of 256), then fp32 activation buffers.
  _Float16* wf16   = (_Float16*)d_ws;
  _Float16* convW  = wf16;                 // 9*64*96 = 55296
  _Float16* pwCR0  = wf16 + 55296;         // 4096
  _Float16* pwCR1  = pwCR0 + 4096;
  _Float16* pwFR0  = pwCR1 + 4096;
  _Float16* pwFR1  = pwFR0 + 4096;
  _Float16* dpW16  = pwFR1 + 4096;         // end = 75776 halfs
  float* ws = (float*)((char*)d_ws + 75776 * sizeof(_Float16));
  size_t o = 0;
  float* proj  = ws + o; o += 64;
  float* stats = ws + o; o += 16;
  float* maps  = ws + o; o += 3 * LHW; o = (o + 63) & ~(size_t)63;
  float* agg   = ws + o; o += (size_t)DN * LHW;
  float* La    = ws + o; o += (size_t)DN * LHW;
  float* Lb    = ws + o; o += (size_t)DN * LHW;
  float* Lc    = ws + o; o += (size_t)DN * LHW;
  float* Fa    = ws + o; o += (size_t)CCH * FHW;
  float* Fb    = ws + o; o += (size_t)CCH * FHW;
  float* Fc    = ws + o; o += (size_t)CCH * FHW;

  const int BT = 256;
  const int lowTot  = DN  * LHW;            // 393216
  const int fullTot = CCH * FHW;            // 6291456
  const int pwBlkL  = (LHW  * 2) / BT;      // one wave per 16-col tile: 48
  const int pwBlkF  = (FHW  * 2) / BT;      // 768
  const int cvBlkF  = (FHW  * 2) / BT;      // 768

  // ---- one-time weight repack to f16 ----
  pack_conv_w_k<<<(9*64*96)/BT, BT, 0, stream>>>(fr_cw, convW);
  pack_w_f16_k<<<16, BT, 0, stream>>>(cr_pw_w,            pwCR0, 4096);
  pack_w_f16_k<<<16, BT, 0, stream>>>(cr_pw_w + 4096,     pwCR1, 4096);
  pack_w_f16_k<<<16, BT, 0, stream>>>(fr_pw_w,            pwFR0, 4096);
  pack_w_f16_k<<<16, BT, 0, stream>>>(fr_pw_w + 4096,     pwFR1, 4096);
  pack_w_f16_k<<<16, BT, 0, stream>>>(dp_w,               dpW16, 4096);

  for (int r = 0; r < NV; ++r) {
    // ---- lowres branch ----
    proj_k<<<1, 32, 0, stream>>>(Kin, Ein, r, proj);
    cost_volume_k<<<lowTot/BT, BT, 0, stream>>>(low, proj, r, agg);

    // cost refine block 1: agg -> Lc
    dwconv3_k<<<lowTot/BT, BT, 0, stream>>>(agg, cr_dw, La, LH, LW, lowTot);
    pw_gemm16_k<<<pwBlkL, BT, 0, stream>>>(pwCR0, La, cr_pw_b, Lb, LHW);
    gn_stats_k<<<8, BT, 0, stream>>>(Lb, stats, 8, LHW);
    gn_gelu_res_k<<<lowTot/BT, BT, 0, stream>>>(Lb, stats, cr_gn_g, cr_gn_b,
                                                agg, Lc, LHW, 8, lowTot);
    // cost refine block 2: Lc -> La
    dwconv3_k<<<lowTot/BT, BT, 0, stream>>>(Lc, cr_dw + DN*9, La, LH, LW, lowTot);
    pw_gemm16_k<<<pwBlkL, BT, 0, stream>>>(pwCR1, La, cr_pw_b + DN, Lb, LHW);
    gn_stats_k<<<8, BT, 0, stream>>>(Lb, stats, 8, LHW);
    gn_gelu_res_k<<<lowTot/BT, BT, 0, stream>>>(Lb, stats, cr_gn_g + DN, cr_gn_b + DN,
                                                Lc, La, LHW, 8, lowTot);
    // logits = dp_w @ ref_feat + dp_b - agg - rc(La)  -> Lb
    pw_logits16_k<<<pwBlkL, BT, 0, stream>>>(dpW16, low + (size_t)r*CCH*LHW, dp_b, Lb,
                                             agg, La, LHW);
    softmax_stats_k<<<LHW/BT, BT, 0, stream>>>(Lb, maps);

    // ---- fullres branch ----  (resize(images) at FHxFW is identity)
    conv3x3_gemm16_k<<<cvBlkF, BT, 0, stream>>>(full + (size_t)r*CCH*FHW,
                                                imgs + (size_t)r*3*FHW, convW, fr_cb, Fa);
    // fr block 1: Fa -> Fb
    dwconv3_k<<<fullTot/BT, BT, 0, stream>>>(Fa, fr_dw, Fb, FH, FW, fullTot);
    pw_gemm16_k<<<pwBlkF, BT, 0, stream>>>(pwFR0, Fb, fr_pw_b, Fc, FHW);
    gn_stats_k<<<8, BT, 0, stream>>>(Fc, stats, 8, FHW);
    gn_gelu_res_k<<<fullTot/BT, BT, 0, stream>>>(Fc, stats, fr_gn_g, fr_gn_b,
                                                 Fa, Fb, FHW, 8, fullTot);
    // fr block 2: Fb -> Fa
    dwconv3_k<<<fullTot/BT, BT, 0, stream>>>(Fb, fr_dw + CCH*9, Fa, FH, FW, fullTot);
    pw_gemm16_k<<<pwBlkF, BT, 0, stream>>>(pwFR1, Fa, fr_pw_b + CCH, Fc, FHW);
    gn_stats_k<<<8, BT, 0, stream>>>(Fc, stats, 8, FHW);
    gn_gelu_res_k<<<fullTot/BT, BT, 0, stream>>>(Fc, stats, fr_gn_g + CCH, fr_gn_b + CCH,
                                                 Fb, Fa, FHW, 8, fullTot);
    // heads
    heads_k<<<FHW/BT, BT, 0, stream>>>(Fa, maps, res_w, res_b, conf_w, conf_b, outp, r);
  }
}
